// ExpertChoiceMoELayer_64003602645071
// MI455X (gfx1250) — matrix operand order repaired and checked by reading, not compiled, not run on hardware
//
#include <hip/hip_runtime.h>
#include <hip/hip_bf16.h>

typedef __attribute__((ext_vector_type(16))) __bf16 v16bf;
typedef __attribute__((ext_vector_type(8)))  __bf16 v8bf;
typedef __attribute__((ext_vector_type(8)))  float  v8f;

#define B_ 4
#define S_ 2048
#define D_ 1024
#define E_ 8
#define F_ 4096
#define C_ 512

static __device__ __forceinline__ unsigned short f2bf(float f) {
    union { __hip_bfloat16 h; unsigned short u; } cv;
    cv.h = __float2bfloat16(f);
    return cv.u;
}

// ---------------------------------------------------------------------------
// Kernel 1: router logits + softmax over E, plus x -> bf16 conversion.
// One wave32 per token, 8 waves / block.
// ---------------------------------------------------------------------------
__global__ void moe_router_kernel(const float* __restrict__ x,
                                  const float* __restrict__ router,
                                  float* __restrict__ aff,
                                  unsigned short* __restrict__ xbf) {
    int wave = threadIdx.x >> 5;
    int lane = threadIdx.x & 31;
    int tok  = blockIdx.x * 8 + wave;            // 0 .. B*S-1
    const float* xr = x + (size_t)tok * D_;

    float acc[E_];
#pragma unroll
    for (int e = 0; e < E_; ++e) acc[e] = 0.0f;

    for (int d = lane; d < D_; d += 32) {
        float xv = xr[d];
        xbf[(size_t)tok * D_ + d] = f2bf(xv);
#pragma unroll
        for (int e = 0; e < E_; ++e) acc[e] += xv * router[d * E_ + e];
    }
#pragma unroll
    for (int e = 0; e < E_; ++e) {
#pragma unroll
        for (int off = 16; off > 0; off >>= 1)
            acc[e] += __shfl_xor(acc[e], off, 32);
    }
    float m = acc[0];
#pragma unroll
    for (int e = 1; e < E_; ++e) m = fmaxf(m, acc[e]);
    float p[E_]; float sum = 0.0f;
#pragma unroll
    for (int e = 0; e < E_; ++e) { p[e] = expf(acc[e] - m); sum += p[e]; }
    float inv = 1.0f / sum;
    if (lane < E_) aff[(size_t)tok * E_ + lane] = p[lane] * inv;
}

// ---------------------------------------------------------------------------
// Kernel 2: fused fp32 -> bf16 convert + transpose of weights.
//   in : w  (E, K, N) fp32
//   out: o  (E, N, K) bf16   (so GEMM B-tiles are K-contiguous per column)
// ---------------------------------------------------------------------------
__global__ void moe_cvt_t_kernel(const float* __restrict__ w,
                                 unsigned short* __restrict__ o,
                                 int K, int N) {
    __shared__ unsigned short T[64 * 72];   // [n][k], row stride 72
    int e  = blockIdx.z;
    int k0 = blockIdx.y * 64;
    int n0 = blockIdx.x * 64;
    int t  = threadIdx.x;

    int kl  = t >> 2;             // 0..63
    int nl4 = (t & 3) * 16;       // 0,16,32,48
    const float* wp = w + ((size_t)e * K + k0 + kl) * N + n0 + nl4;
#pragma unroll
    for (int j = 0; j < 4; ++j) {
        float4 v = ((const float4*)wp)[j];
        int nn = nl4 + j * 4;
        T[(nn + 0) * 72 + kl] = f2bf(v.x);
        T[(nn + 1) * 72 + kl] = f2bf(v.y);
        T[(nn + 2) * 72 + kl] = f2bf(v.z);
        T[(nn + 3) * 72 + kl] = f2bf(v.w);
    }
    __syncthreads();
    int nl = t >> 2;              // 0..63
    int kc = (t & 3) * 16;        // 0,16,32,48
    unsigned short* op = o + ((size_t)e * N + n0 + nl) * K + k0 + kc;
    *(v8bf*)op       = *(const v8bf*)&T[nl * 72 + kc];
    *(v8bf*)(op + 8) = *(const v8bf*)&T[nl * 72 + kc + 8];
}

// ---------------------------------------------------------------------------
// Kernel 3: expert-choice top-C per (b,e) via rank counting (stable, matches
// lax.top_k: descending value, ties broken by lower index).
// ---------------------------------------------------------------------------
__global__ void moe_topk_kernel(const float* __restrict__ aff,
                                int* __restrict__ idx,
                                float* __restrict__ gate) {
    __shared__ float sv[S_];
    int be = blockIdx.x;                  // b*E + e
    int b = be >> 3, e = be & 7;
    for (int s = threadIdx.x; s < S_; s += 256)
        sv[s] = aff[((size_t)b * S_ + s) * E_ + e];
    __syncthreads();
    for (int s = threadIdx.x; s < S_; s += 256) {
        float v = sv[s];
        int rank = 0;
        for (int j = 0; j < S_; ++j) {
            float u = sv[j];
            rank += (u > v) || (u == v && j < s);
        }
        if (rank < C_) {
            idx [(size_t)be * C_ + rank] = s;
            gate[(size_t)be * C_ + rank] = v;
        }
    }
}

// ---------------------------------------------------------------------------
// Kernel 4: tiled bf16 WMMA grouped GEMM, block tile 128(M) x 128(N),
// 256 threads = 8 waves (4x2). Each wave computes a 32x64 tile:
//   2 A fragments x 4 B fragments = eight 16x16x32 WMMAs per K-step,
//   every A frag reused 4x, every B frag reused 2x.
//   mode 1: A = gathered x_bf16 rows (idx), Wt = w1^T, out = gelu(.+b1) -> h
//   mode 2: A = h rows,                Wt = w2^T, out = (.+b2)*gate scatter-add
// Wt layout: (E, N, K) bf16 -> B-tile loads are contiguous 32B per thread.
// ---------------------------------------------------------------------------
__global__ void moe_gemm_wmma(const unsigned short* __restrict__ Abase,
                              const unsigned short* __restrict__ Wtbase,
                              const float* __restrict__ bias,
                              const int* __restrict__ idx,
                              const float* __restrict__ gate,
                              unsigned short* __restrict__ Hout,
                              float* __restrict__ Out,
                              int K, int N, int mode) {
    __shared__ __bf16 As[128 * 40];   // [m][k], row stride 40 (pad)
    __shared__ __bf16 Bs[128 * 40];   // [n][k], row stride 40 (pad)

    int be = blockIdx.z; int b = be >> 3; int e = be & 7;
    int n0 = blockIdx.x * 128;
    int m0 = blockIdx.y * 128;
    int t = threadIdx.x;
    int lane = t & 31, wave = t >> 5;
    int wrow = wave >> 1, wcol = wave & 1;    // 4 x 2 wave grid
    int lm = lane & 15, half = lane >> 4;

    const unsigned short* Wt = Wtbase + (size_t)e * N * K;

    // A-tile loader: 128 rows x 32 k, 16 bf16 (two v8bf) per thread
    int ar = t >> 1;
    int akc = (t & 1) * 16;
    const unsigned short* aRow;
    if (mode == 1) {
        int srow = idx[(size_t)be * C_ + (m0 + ar)];
        aRow = Abase + ((size_t)b * S_ + srow) * K + akc;
    } else {
        aRow = Abase + ((size_t)be * C_ + (m0 + ar)) * K + akc;
    }
    // B-tile loader: 128 n-rows x 32 k, 16 bf16 (two v8bf) per thread
    int bn_l = t >> 1;
    int bk_l = (t & 1) * 16;
    const unsigned short* bRow = Wt + (size_t)(n0 + bn_l) * K + bk_l;

    v8f acc[2][4];
#pragma unroll
    for (int mi = 0; mi < 2; ++mi)
#pragma unroll
        for (int sub = 0; sub < 4; ++sub) acc[mi][sub] = (v8f){};

    for (int k0 = 0; k0 < K; k0 += 32) {
        // stage A (row-major [m][k])
        *(v8bf*)(&As[ar * 40 + akc])     = *(const v8bf*)(aRow + k0);
        *(v8bf*)(&As[ar * 40 + akc + 8]) = *(const v8bf*)(aRow + k0 + 8);
        // stage B ([n][k], K-contiguous in global)
        *(v8bf*)(&Bs[bn_l * 40 + bk_l])     = *(const v8bf*)(bRow + k0);
        *(v8bf*)(&Bs[bn_l * 40 + bk_l + 8]) = *(const v8bf*)(bRow + k0 + 8);
        __syncthreads();

        // two A fragments: lane(half h) holds K = {8h..8h+7} then {16+8h..}
        v16bf a[2];
#pragma unroll
        for (int mi = 0; mi < 2; ++mi) {
            int am = wrow * 32 + mi * 16 + lm;
            v8bf alo = *(const v8bf*)(&As[am * 40 + 8 * half]);
            v8bf ahi = *(const v8bf*)(&As[am * 40 + 16 + 8 * half]);
#pragma unroll
            for (int i = 0; i < 8; ++i) { a[mi][i] = alo[i]; a[mi][8 + i] = ahi[i]; }
        }

#pragma unroll
        for (int sub = 0; sub < 4; ++sub) {
            int bn = wcol * 64 + sub * 16 + lm;
            v8bf blo = *(const v8bf*)(&Bs[bn * 40 + 8 * half]);
            v8bf bhi = *(const v8bf*)(&Bs[bn * 40 + 16 + 8 * half]);
            v16bf bb;
#pragma unroll
            for (int i = 0; i < 8; ++i) { bb[i] = blo[i]; bb[8 + i] = bhi[i]; }
#pragma unroll
            for (int mi = 0; mi < 2; ++mi) {
                acc[mi][sub] = __builtin_amdgcn_wmma_f32_16x16x32_bf16(
                    false, a[mi], false, bb, (short)0, acc[mi][sub], false, false);
            }
        }
        __syncthreads();
    }

    // epilogue: accumulator element r -> row (wrow*32 + mi*16 + r + 8*half)
#pragma unroll
    for (int sub = 0; sub < 4; ++sub) {
        int col = n0 + wcol * 64 + sub * 16 + lm;
        float bv = bias[(size_t)e * N + col];
#pragma unroll
        for (int mi = 0; mi < 2; ++mi) {
            if (mode == 1) {
#pragma unroll
                for (int r = 0; r < 8; ++r) {
                    int row = m0 + wrow * 32 + mi * 16 + r + 8 * half;
                    float v = acc[mi][sub][r] + bv;
                    v = 0.5f * v * (1.0f + erff(v * 0.70710678118654752f));
                    Hout[((size_t)be * C_ + row) * N + col] = f2bf(v);
                }
            } else {
#pragma unroll
                for (int r = 0; r < 8; ++r) {
                    int row = m0 + wrow * 32 + mi * 16 + r + 8 * half;
                    float g = gate[(size_t)be * C_ + row];
                    int s   = idx [(size_t)be * C_ + row];
                    atomicAdd(Out + ((size_t)b * S_ + s) * N + col,
                              (acc[mi][sub][r] + bv) * g);
                }
            }
        }
    }
}

// ---------------------------------------------------------------------------
extern "C" void kernel_launch(void* const* d_in, const int* in_sizes, int n_in,
                              void* d_out, int out_size, void* d_ws, size_t ws_size,
                              hipStream_t stream) {
    (void)in_sizes; (void)n_in; (void)ws_size;
    const float* x      = (const float*)d_in[0];
    const float* router = (const float*)d_in[1];
    const float* w1     = (const float*)d_in[2];
    const float* b1     = (const float*)d_in[3];
    const float* w2     = (const float*)d_in[4];
    const float* b2     = (const float*)d_in[5];
    float* out = (float*)d_out;

    char* ws = (char*)d_ws;
    size_t off = 0;
    auto alloc = [&](size_t bytes) -> void* {
        void* p = ws + off;
        off = (off + bytes + 255) & ~(size_t)255;
        return p;
    };
    float*          aff  = (float*)         alloc((size_t)B_ * S_ * E_ * 4);
    int*            idx  = (int*)           alloc((size_t)B_ * E_ * C_ * 4);
    float*          gate = (float*)         alloc((size_t)B_ * E_ * C_ * 4);
    unsigned short* xbf  = (unsigned short*)alloc((size_t)B_ * S_ * D_ * 2);
    unsigned short* w1t  = (unsigned short*)alloc((size_t)E_ * D_ * F_ * 2); // (E,F,D)
    unsigned short* w2t  = (unsigned short*)alloc((size_t)E_ * F_ * D_ * 2); // (E,D,F)
    unsigned short* h    = (unsigned short*)alloc((size_t)B_ * E_ * C_ * F_ * 2);

    hipMemsetAsync(d_out, 0, (size_t)out_size * sizeof(float), stream);

    moe_router_kernel<<<(B_ * S_) / 8, 256, 0, stream>>>(x, router, aff, xbf);
    // w1 (E, D, F) -> w1t (E, F, D)
    moe_cvt_t_kernel<<<dim3(F_ / 64, D_ / 64, E_), 256, 0, stream>>>(w1, w1t, D_, F_);
    // w2 (E, F, D) -> w2t (E, D, F)
    moe_cvt_t_kernel<<<dim3(D_ / 64, F_ / 64, E_), 256, 0, stream>>>(w2, w2t, F_, D_);
    moe_topk_kernel<<<B_ * E_, 256, 0, stream>>>(aff, idx, gate);

    // Layer 1: gathered tokens x w1 -> gelu -> h (bf16)
    moe_gemm_wmma<<<dim3(F_ / 128, C_ / 128, B_ * E_), 256, 0, stream>>>(
        xbf, w1t, b1, idx, gate, h, out, D_, F_, 1);
    // Layer 2: h x w2 -> (+b2)*gate -> atomic scatter into out
    moe_gemm_wmma<<<dim3(D_ / 128, C_ / 128, B_ * E_), 256, 0, stream>>>(
        h, w2t, b2, idx, gate, nullptr, out, F_, D_, 2);
}
